// ColbertLoss_59416577572864
// MI455X (gfx1250) — matrix operand order repaired and checked by reading, not compiled, not run on hardware
//
#include <hip/hip_runtime.h>
#include <hip/hip_bf16.h>
#include <math.h>

typedef __attribute__((ext_vector_type(16))) _Float16 v16h;
typedef __attribute__((ext_vector_type(8)))  _Float16 v8h;
typedef __attribute__((ext_vector_type(4)))  _Float16 v4h;
typedef __attribute__((ext_vector_type(8)))  float    v8f;

#define NB 32      // batch (queries)
#define NC 512     // docs
#define NS 32      // query tokens
#define ND 128     // doc tokens
#define NH 128     // hidden
#define INV_TEMP 50.0f  // 1/0.02

#define QELEMS (NB * NS * NH)   // 131072

// ---------------------------------------------------------------------------
// Pre-pass: split Q (f32) into hi/lo f16 planes, row-major in K.
// Bandwidth-bound, runs once: 512 KB in, 512 KB out.
// ---------------------------------------------------------------------------
__global__ __launch_bounds__(256) void
convert_q_kernel(const float* __restrict__ Q,
                 _Float16* __restrict__ Qhi, _Float16* __restrict__ Qlo)
{
    const int idx = (blockIdx.x * 256 + threadIdx.x) * 4;  // float4 granularity
    float4 x = *(const float4*)(Q + idx);
    float xs[4] = {x.x, x.y, x.z, x.w};
    v4h h, l;
#pragma unroll
    for (int i = 0; i < 4; ++i) {
        _Float16 hv = (_Float16)xs[i];
        h[i] = hv;
        l[i] = (_Float16)(xs[i] - (float)hv);
    }
    *(v4h*)(Qhi + idx) = h;
    *(v4h*)(Qlo + idx) = l;
}

// Split 8 consecutive f32 into hi/lo f16, writing frag elems [base, base+8).
__device__ __forceinline__ void cvt_chunk8(const float* __restrict__ src,
                                           v16h& hi, v16h& lo, int base) {
    float4 a = *(const float4*)(src);
    float4 b = *(const float4*)(src + 4);
    float xs[8] = {a.x, a.y, a.z, a.w, b.x, b.y, b.z, b.w};
#pragma unroll
    for (int i = 0; i < 8; ++i) {
        _Float16 h = (_Float16)xs[i];
        hi[base + i] = h;
        lo[base + i] = (_Float16)(xs[i] - (float)h);
    }
}

__device__ __forceinline__ v16h join8(v8h a, v8h b) {
    return __builtin_shufflevector(a, b, 0, 1, 2, 3, 4, 5, 6, 7,
                                         8, 9, 10, 11, 12, 13, 14, 15);
}

// B-operand fragments for one s-tile: Q rows, ISA layout per lane
// (lanes 0-15: K k0..k0+7,k0+16..k0+23; lanes 16-31: K k0+8..k0+15,k0+24..k0+31).
struct QFrag { v16h h[4]; v16h l[4]; };

__device__ __forceinline__ void load_qfrags(const _Float16* __restrict__ Qhi,
                                            const _Float16* __restrict__ Qlo,
                                            int off, QFrag& f) {
#pragma unroll
    for (int ks = 0; ks < 4; ++ks) {
        const int o = off + ks * 32;
        f.h[ks] = join8(*(const v8h*)(Qhi + o), *(const v8h*)(Qhi + o + 16));
        f.l[ks] = join8(*(const v8h*)(Qlo + o), *(const v8h*)(Qlo + o + 16));
    }
}

// ---------------------------------------------------------------------------
// One block per doc c; 8 waves; wave w owns d-tile w (16 doc tokens).
// Tile orientation: D[m=d][n=s] = P_tile x Q_tile^T so the MaxSim max-over-d
// is 7 in-lane max ops + one half-swap shuffle.
// A operand (stationary, converted once): P rows, ISA A-layout
//   lane<16 : elems0-7 = K o..o+7,   elems8-15 = K o+16..o+23
//   lane>=16: elems0-7 = K o+8..o+15, elems8-15 = K o+24..o+31
// B operand (streamed, pre-converted): Q rows, same ISA K layout as A.
// ---------------------------------------------------------------------------
__global__ __launch_bounds__(256) void
colbert_scores_kernel(const _Float16* __restrict__ Qhi,  // [NB][NS][NH]
                      const _Float16* __restrict__ Qlo,  // [NB][NS][NH]
                      const float* __restrict__ P,       // [NC][ND][NH]
                      float* __restrict__ S)             // [NB][NC]
{
    const int c    = blockIdx.x;
    const int wave = threadIdx.x >> 5;      // d-tile index 0..7
    const int lane = threadIdx.x & 31;
    const int r    = lane & 15;             // row (A) / column (B) within tile
    const int ksel = (lane >> 4) * 8;       // A-fragment K sub-chunk
    const int bsel = (lane >> 4) * 8;       // B-fragment K sub-chunk

    __shared__ float lpartial[NB][8][NS];   // [b][wave][s] tile row-max (32 KB)

    // ---- Split-convert A fragments (P rows) for all 4 K-steps; keep in regs.
    const float* prow = P + (((size_t)c * ND) + wave * 16 + r) * NH;
    v16h Ph[4], Pl[4];
#pragma unroll
    for (int ks = 0; ks < 4; ++ks) {
        const int o = ks * 32 + ksel;
        cvt_chunk8(prow + o,      Ph[ks], Pl[ks], 0);
        cvt_chunk8(prow + o + 16, Ph[ks], Pl[ks], 8);
    }

    // Q fragment base offset for (b, st): ((b*NS + st*16 + r) * NH) + bsel
    QFrag f0, f1;
    load_qfrags(Qhi, Qlo, (r * NH) + bsel, f0);   // (b=0, st=0)

    for (int b = 0; b < NB; ++b) {
        const int base = (b * NS + r) * NH + bsel;
        // prefetch s-tile 1 of this b
        load_qfrags(Qhi, Qlo, base + 16 * NH, f1);

        // ---- s-tile 0 ----
        v8f acc = {};
#pragma unroll
        for (int ks = 0; ks < 4; ++ks) {
            acc = __builtin_amdgcn_wmma_f32_16x16x32_f16(
                      false, Ph[ks], false, f0.h[ks], (short)0, acc, false, false);
            acc = __builtin_amdgcn_wmma_f32_16x16x32_f16(
                      false, Ph[ks], false, f0.l[ks], (short)0, acc, false, false);
            acc = __builtin_amdgcn_wmma_f32_16x16x32_f16(
                      false, Pl[ks], false, f0.h[ks], (short)0, acc, false, false);
        }
        float rm0 = acc[0];
#pragma unroll
        for (int i = 1; i < 8; ++i) rm0 = fmaxf(rm0, acc[i]);   // max over m-half
        rm0 = fmaxf(rm0, __shfl_xor(rm0, 16, 32));              // merge halves

        // prefetch s-tile 0 of next b (dummy reload of b on last iter)
        const int bn = (b + 1 < NB) ? (b + 1) : b;
        load_qfrags(Qhi, Qlo, (bn * NS + r) * NH + bsel, f0);

        // ---- s-tile 1 ----
        v8f acc1 = {};
#pragma unroll
        for (int ks = 0; ks < 4; ++ks) {
            acc1 = __builtin_amdgcn_wmma_f32_16x16x32_f16(
                       false, Ph[ks], false, f1.h[ks], (short)0, acc1, false, false);
            acc1 = __builtin_amdgcn_wmma_f32_16x16x32_f16(
                       false, Ph[ks], false, f1.l[ks], (short)0, acc1, false, false);
            acc1 = __builtin_amdgcn_wmma_f32_16x16x32_f16(
                       false, Pl[ks], false, f1.h[ks], (short)0, acc1, false, false);
        }
        float rm1 = acc1[0];
#pragma unroll
        for (int i = 1; i < 8; ++i) rm1 = fmaxf(rm1, acc1[i]);
        rm1 = fmaxf(rm1, __shfl_xor(rm1, 16, 32));

        // lane L holds the value for s = L: st0 for lanes 0-15, st1 for 16-31
        lpartial[b][wave][lane] = (lane < 16) ? rm0 : rm1;
    }

    __syncthreads();

    // Final reduction: wave w handles b = w, w+8, w+16, w+24.
#pragma unroll
    for (int i = 0; i < 4; ++i) {
        const int bb = wave + i * 8;
        float m = lpartial[bb][0][lane];
#pragma unroll
        for (int w2 = 1; w2 < 8; ++w2) m = fmaxf(m, lpartial[bb][w2][lane]);
#pragma unroll
        for (int msk = 1; msk < 32; msk <<= 1) m += __shfl_xor(m, msk, 32);
        if (lane == 0) S[(size_t)bb * NC + c] = m * INV_TEMP;
    }
}

// Single block, 32 waves: wave w computes logsumexp(S[w,:]) - S[w,0]; mean -> out.
__global__ __launch_bounds__(1024) void
colbert_loss_kernel(const float* __restrict__ S, float* __restrict__ out)
{
    const int wave = threadIdx.x >> 5;
    const int lane = threadIdx.x & 31;
    const float* row = S + (size_t)wave * NC;

    float vals[16];
    float m = -INFINITY;
#pragma unroll
    for (int i = 0; i < 16; ++i) {
        vals[i] = row[lane + i * 32];
        m = fmaxf(m, vals[i]);
    }
#pragma unroll
    for (int msk = 1; msk < 32; msk <<= 1) m = fmaxf(m, __shfl_xor(m, msk, 32));

    float ssum = 0.f;
#pragma unroll
    for (int i = 0; i < 16; ++i) ssum += __expf(vals[i] - m);
#pragma unroll
    for (int msk = 1; msk < 32; msk <<= 1) ssum += __shfl_xor(ssum, msk, 32);

    __shared__ float lds[32];
    if (lane == 0) lds[wave] = (m + __logf(ssum)) - row[0];
    __syncthreads();
    if (threadIdx.x == 0) {
        float acc = 0.f;
#pragma unroll
        for (int i = 0; i < 32; ++i) acc += lds[i];
        out[0] = acc / (float)NB;
    }
}

extern "C" void kernel_launch(void* const* d_in, const int* in_sizes, int n_in,
                              void* d_out, int out_size, void* d_ws, size_t ws_size,
                              hipStream_t stream) {
    const float* Q = (const float*)d_in[0];   // [32,32,128] f32
    const float* P = (const float*)d_in[1];   // [512,128,128] f32

    // Workspace layout: [0,64KB) scores, then Q hi plane (256KB), Q lo plane.
    char* ws = (char*)d_ws;
    float*     S   = (float*)ws;                                  // 32*512 f32
    _Float16*  Qhi = (_Float16*)(ws + 64 * 1024);                 // 131072 f16
    _Float16*  Qlo = (_Float16*)(ws + 64 * 1024 + QELEMS * 2);    // 131072 f16

    convert_q_kernel<<<QELEMS / (256 * 4), 256, 0, stream>>>(Q, Qhi, Qlo);
    colbert_scores_kernel<<<NC, 256, 0, stream>>>(Qhi, Qlo, P, S);
    colbert_loss_kernel<<<1, 1024, 0, stream>>>(S, (float*)d_out);
}
